// LearnedDistanceFunction_17660905521625
// MI455X (gfx1250) — matrix operand order repaired and checked by reading, not compiled, not run on hardware
//
#include <hip/hip_runtime.h>
#include <hip/hip_bf16.h>

// ---------------------------------------------------------------------------
// LearnedDistanceFunction for MI455X (gfx1250, wave32, WMMA).
// E=256, H=8, HD=32, P=96, Lp=16, S=128, Ls=16.
//
// Fused math:  sim[p,s] = sum_h sum_j abar[p,s,h,j] * vu[s,h,j] + c
//   abar  = column mean of softmax(q k^T / sqrt(HD))   (16x16 per head)
//   vu    = sat @ (Wv^T u) + (bv . u_h),  u = Wout^T w, c = w.bout + b
// This removes the 25.8-GFLOP ctx@Wout path entirely.
// ---------------------------------------------------------------------------

typedef __attribute__((ext_vector_type(16))) __bf16 v16bf;
typedef __attribute__((ext_vector_type(8)))  float  v8f;

#define E   256
#define H   8
#define HD  32
#define NP  96
#define NS  128
#define L   16
#define QSCALE 0.17677669529663687f   // 1/sqrt(32)

// --------------------------------------------------------------------------
// fold_kernel: u[e] = sum_e' w[e'] Wout[e',e];  wvu[h,e] = sum_d Wv[h*32+d,e]u[h*32+d]
//              bvu[h] = sum_d bv[h*32+d] u[h*32+d];  c = w.bout + b
// --------------------------------------------------------------------------
__global__ __launch_bounds__(256) void fold_kernel(
    const float* __restrict__ ipw, const float* __restrict__ ipb,
    const float* __restrict__ opw, const float* __restrict__ opb,
    const float* __restrict__ fw,  const float* __restrict__ fb,
    float* __restrict__ wvu, float* __restrict__ bvu_c)
{
    __shared__ float us[E];
    const int e = threadIdx.x;
    float acc = 0.f;
    for (int ep = 0; ep < E; ++ep) acc += fw[ep] * opw[ep * E + e];
    us[e] = acc;
    __syncthreads();
    for (int h = 0; h < H; ++h) {
        float a = 0.f;
        for (int d = 0; d < HD; ++d)
            a += ipw[(2 * E + h * HD + d) * E + e] * us[h * HD + d];
        wvu[h * E + e] = a;
    }
    if (e < H) {
        float a = 0.f;
        for (int d = 0; d < HD; ++d) a += ipb[2 * E + e * HD + d] * us[e * HD + d];
        bvu_c[e] = a;
    }
    if (e == H) {
        float a = fb[0];
        for (int ep = 0; ep < E; ++ep) a += fw[ep] * opb[ep];
        bvu_c[H] = a;
    }
}

// --------------------------------------------------------------------------
// vu_kernel: vu[(s*16+j)*8+h] = sat[s,j,:] . wvu[h,:] + bvu[h]
// --------------------------------------------------------------------------
__global__ __launch_bounds__(256) void vu_kernel(
    const float* __restrict__ sat, const float* __restrict__ wvu,
    const float* __restrict__ bvu_c, float* __restrict__ vu)
{
    const int tid = blockIdx.x * blockDim.x + threadIdx.x;   // 16384
    const int h = tid & 7, sj = tid >> 3;
    const float* row = sat + sj * E;
    const float* wr  = wvu + h * E;
    float acc = bvu_c[h];
    for (int e = 0; e < E; ++e) acc += row[e] * wr[e];
    vu[tid] = acc;
}

// --------------------------------------------------------------------------
// proj_kernel: Y[m,n] = (X[m,:] . W[n,:] + bias[n]) * scale   via bf16 WMMA.
// One wave per 16x16 output tile, K loop 8 x (16x16x32).
// A layout: lane<16 M=l, K=0..15; lane>=16 M=l, K=16..31.
// B layout: lane<16 N=l K=0..15, else K=16..31.
// --------------------------------------------------------------------------
__global__ __launch_bounds__(256) void proj_kernel(
    const float* __restrict__ X, const float* __restrict__ W,
    const float* __restrict__ bias, float* __restrict__ Y,
    int nrows, float scale)
{
    const int wid  = (blockIdx.x * blockDim.x + threadIdx.x) >> 5;
    const int lane = threadIdx.x & 31;
    const int tm = wid >> 4, tn = wid & 15;          // 16 tiles along N (=256/16)
    if (tm * 16 >= nrows) return;
    const int half = lane >> 4, l = lane & 15;
    const float* Xrow = X + (tm * 16 + l) * E;       // A-matrix source row
    const float* Wrow = W + (tn * 16 + l) * E;       // B[K,n] = W[n,K]
    const int d0  = half * 8;
    const int bk0 = half * 16;
    v8f c = {};
#pragma unroll
    for (int kk = 0; kk < 8; ++kk) {
        v16bf a, b;
#pragma unroll
        for (int e = 0; e < 16; ++e)
            a[e] = (__bf16)Xrow[kk * 32 + bk0 + e];
#pragma unroll
        for (int e = 0; e < 16; ++e)
            b[e] = (__bf16)Wrow[kk * 32 + bk0 + e];
        c = __builtin_amdgcn_wmma_f32_16x16x32_bf16(false, a, false, b,
                                                    (short)0, c, false, false);
    }
    const int n = tn * 16 + l;
    const float bn = bias[n];
#pragma unroll
    for (int r = 0; r < 8; ++r) {
        const int m = tm * 16 + r + half * 8;        // C layout: M = r + 8*(lane/16)
        Y[m * E + n] = (c[r] + bn) * scale;
    }
}

// --------------------------------------------------------------------------
// repack_kernel: gather per-lane WMMA operand layout, cvt f32->bf16, store
// 32 B/lane so attn's loads are two GLOBAL_LOAD_B128 per operand.
// mode 0: A layout from q (tiles = P*H).  mode 1: B layout (k^T) (tiles = S*H).
// --------------------------------------------------------------------------
__global__ __launch_bounds__(256) void repack_kernel(
    const float* __restrict__ src, __bf16* __restrict__ dst, int ntiles, int mode)
{
    const int wid  = (blockIdx.x * blockDim.x + threadIdx.x) >> 5;
    if (wid >= ntiles) return;
    const int lane = threadIdx.x & 31;
    const int outer = wid >> 3, h = wid & 7;
    const int half = lane >> 4, l = lane & 15;
    v16bf v;
    if (mode == 0) {
        const float* row = src + (outer * 16 + l) * E + h * HD;
        const int d0 = half * 16;
#pragma unroll
        for (int e = 0; e < 16; ++e)
            v[e] = (__bf16)row[d0 + e];
    } else {
        const float* row = src + (outer * 16 + l) * E + h * HD + half * 16;
#pragma unroll
        for (int e = 0; e < 16; ++e) v[e] = (__bf16)row[e];
    }
    reinterpret_cast<v16bf*>(dst)[wid * 32 + lane] = v;
}

// --------------------------------------------------------------------------
// attn_kernel: one wave per (p,s) pair. Per head: 1 WMMA (16x16x32 bf16)
// -> scores, shuffle softmax over j (16 lanes/half), column-mean,
// dot with vu[s,h,:].  sim = acc/16 + c.
// --------------------------------------------------------------------------
__global__ __launch_bounds__(256) void attn_kernel(
    const __bf16* __restrict__ qpack, const __bf16* __restrict__ kpack,
    const float* __restrict__ vu, const float* __restrict__ bvu_c,
    float* __restrict__ out)
{
    const int wid  = (blockIdx.x * blockDim.x + threadIdx.x) >> 5;
    const int lane = threadIdx.x & 31;
    const int p = wid >> 7;          // / NS
    const int s = wid & 127;         // % NS
    const int j = lane & 15;
    const v16bf* qa = reinterpret_cast<const v16bf*>(qpack);
    const v16bf* kb = reinterpret_cast<const v16bf*>(kpack);
    float acc = 0.f;
#pragma unroll
    for (int h = 0; h < H; ++h) {
        v16bf a = qa[(p * H + h) * 32 + lane];
        v16bf b = kb[(s * H + h) * 32 + lane];
        v8f c = {};
        c = __builtin_amdgcn_wmma_f32_16x16x32_bf16(false, a, false, b,
                                                    (short)0, c, false, false);
        const float vuj = vu[(s * 16 + j) * 8 + h];
        float colsum = 0.f;
#pragma unroll
        for (int r = 0; r < 8; ++r) {
            float sv = c[r];                     // row M=r+8*half, col N=j
            float m = sv;
            m = fmaxf(m, __shfl_xor(m, 1));
            m = fmaxf(m, __shfl_xor(m, 2));
            m = fmaxf(m, __shfl_xor(m, 4));
            m = fmaxf(m, __shfl_xor(m, 8));      // row max (within 16-lane half)
            float ev = __expf(sv - m);
            float z = ev;
            z += __shfl_xor(z, 1);
            z += __shfl_xor(z, 2);
            z += __shfl_xor(z, 4);
            z += __shfl_xor(z, 8);               // row sum
            colsum += ev * (1.0f / z);           // attn prob, accumulate column sum
        }
        colsum += __shfl_xor(colsum, 16);        // merge M=0..7 and M=8..15 halves
        float part = colsum * vuj;
        part += __shfl_xor(part, 1);
        part += __shfl_xor(part, 2);
        part += __shfl_xor(part, 4);
        part += __shfl_xor(part, 8);             // sum over j
        acc += part;
    }
    if (lane == 0)
        out[p * NS + s] = acc * (1.0f / 16.0f) + bvu_c[H];
}

// --------------------------------------------------------------------------
extern "C" void kernel_launch(void* const* d_in, const int* in_sizes, int n_in,
                              void* d_out, int out_size, void* d_ws, size_t ws_size,
                              hipStream_t stream) {
    const float* sat  = (const float*)d_in[0];   // [128,16,256]
    const float* pano = (const float*)d_in[1];   // [96,16,256]
    const float* ipw  = (const float*)d_in[2];   // [768,256]
    const float* ipb  = (const float*)d_in[3];   // [768]
    const float* opw  = (const float*)d_in[4];   // [256,256]
    const float* opb  = (const float*)d_in[5];   // [256]
    const float* fw   = (const float*)d_in[6];   // [1,256]
    const float* fb   = (const float*)d_in[7];   // [1]
    float* out = (float*)d_out;                  // [96,128]

    // workspace carve-up (floats then bf16; bf16 regions 32B aligned)
    float* q     = (float*)d_ws;           // 1536*256
    float* k     = q + NP * L * E;         // 2048*256
    float* wvu   = k + NS * L * E;         // 8*256
    float* bvu_c = wvu + H * E;            // 9 (+pad to 16)
    float* vu    = bvu_c + 16;             // 2048*8
    __bf16* qpack = (__bf16*)(vu + NS * L * H);   // 768 tiles * 512 bf16
    __bf16* kpack = qpack + NP * H * 512;         // 1024 tiles * 512 bf16

    fold_kernel<<<1, 256, 0, stream>>>(ipw, ipb, opw, opb, fw, fb, wvu, bvu_c);
    vu_kernel<<<(NS * L * H) / 256, 256, 0, stream>>>(sat, wvu, bvu_c, vu);

    // Q projection (rows=1536, pre-scaled by 1/sqrt(HD)); K projection (rows=2048)
    proj_kernel<<<(NP * L / 16) * 16 / 8, 256, 0, stream>>>(
        pano, ipw, ipb, q, NP * L, QSCALE);
    proj_kernel<<<(NS * L / 16) * 16 / 8, 256, 0, stream>>>(
        sat, ipw + E * E, ipb + E, k, NS * L, 1.0f);

    repack_kernel<<<NP * H / 8, 256, 0, stream>>>(q, qpack, NP * H, 0);
    repack_kernel<<<NS * H / 8, 256, 0, stream>>>(k, kpack, NS * H, 1);

    attn_kernel<<<NP * NS / 8, 256, 0, stream>>>(qpack, kpack, vu, bvu_c, out);
}
